// PointNetAbstraction_59674275610655
// MI455X (gfx1250) — compile-verified, matrix operand order checked by reading, not run on hardware
//
#include <hip/hip_runtime.h>
#include <hip/hip_bf16.h>
#include <stdint.h>

// ---------------------------------------------------------------------------
// CDNA5 (gfx1250) PointNet++ MSG forward.
// Channel-GEMMs on V_WMMA_F32_16X16X32_BF16 (wave32, f32 accum).
//  - Weights pre-padded to [Mpad x Kpad] bf16 -> branchless b128 staging.
//  - Activations K-padded with zero rows -> no guards in the K loop.
//  - Fragments fed by ds_load_b128 (two 16B runs per lane, per ISA layout);
//    X staged transposed in LDS. 80B LDS row stride: aligned, conflict-free.
// ---------------------------------------------------------------------------

typedef __attribute__((ext_vector_type(16))) __bf16 v16bf;
typedef __attribute__((ext_vector_type(8)))  float  v8f;
typedef __attribute__((ext_vector_type(8)))  unsigned short u16x8;
struct BF16Frag { u16x8 lo, hi; };

#define GS(e, tot) for (long long e = (long long)blockIdx.x * blockDim.x + threadIdx.x; \
                        e < (tot); e += (long long)gridDim.x * blockDim.x)

__device__ __forceinline__ unsigned short f32_to_bf16_bits(float f) {
  unsigned int u = __float_as_uint(f);
  u += 0x7fffu + ((u >> 16) & 1u);           // round-to-nearest-even
  return (unsigned short)(u >> 16);
}

static inline int pad32(int x) { return (x + 31) & ~31; }
static inline int pad64(int x) { return (x + 63) & ~63; }

// -------------------------------- misc elementwise -------------------------

__global__ void zero_u16_kernel(unsigned short* __restrict__ p, long long tot) {
  GS(e, tot) p[e] = 0;
}

__global__ void copy_f32_kernel(const float* __restrict__ in, float* __restrict__ out,
                                long long tot) {
  GS(e, tot) out[e] = in[e];
}

// weight prep: fp32 [M,K] -> bf16 [Mpad,Kpad], zero-padded
__global__ void w_prep_kernel(const float* __restrict__ W, int M, int Kdim, int Kpad,
                              unsigned short* __restrict__ out, long long tot) {
  GS(e, tot) {
    int k = (int)(e % Kpad);
    int m = (int)(e / Kpad);
    out[e] = (m < M && k < Kdim) ? f32_to_bf16_bits(W[(size_t)m * Kdim + k])
                                 : (unsigned short)0;
  }
}

// slice channels [B,Cin,n] -> [B,cnum,n] starting at c0
__global__ void slice_channels_kernel(const float* __restrict__ in, int B_, int Cin, int n,
                                      int c0, int cnum, float* __restrict__ out) {
  long long tot = (long long)B_ * cnum * n;
  GS(e, tot) {
    int nn = (int)(e % n);
    long long t = e / n;
    int d = (int)(t % cnum);
    int b = (int)(t / cnum);
    out[e] = in[((size_t)b * Cin + c0 + d) * n + nn];
  }
}

// rows [M, B*n] (f32) -> [B, M, n]
__global__ void rows_to_bcs_kernel(const float* __restrict__ rows, int M, int B_, int n,
                                   float* __restrict__ out) {
  long long tot = (long long)B_ * M * n;
  GS(e, tot) {
    int nn = (int)(e % n);
    long long t = e / n;
    int c = (int)(t % M);
    int b = (int)(t / M);
    out[e] = rows[(size_t)c * ((size_t)B_ * n) + (size_t)b * n + nn];
  }
}

// [B, M, n] (f32) -> rows [M, B*n] (bf16)
__global__ void bcs_to_rows_bf16_kernel(const float* __restrict__ in, int M, int B_, int n,
                                        unsigned short* __restrict__ out) {
  long long cols = (long long)B_ * n;
  long long tot = cols * M;
  GS(e, tot) {
    long long col = e % cols;
    int c = (int)(e / cols);
    int b = (int)(col / n);
    int nn = (int)(col % n);
    out[e] = f32_to_bf16_bits(in[((size_t)b * M + c) * n + nn]);
  }
}

// -------------------------------- FPS --------------------------------------
__global__ void fps_kernel(const float* __restrict__ xyz, int n, int S,
                           int* __restrict__ out_idx) {
  int b = blockIdx.x;
  const float* px = xyz + (size_t)b * 3 * n;      // [3][n]
  extern __shared__ float smem[];
  float* dist = smem;                              // n floats
  float* rmax = smem + n;                          // 256
  int*   rarg = (int*)(rmax + 256);                // 256
  int t = threadIdx.x;
  for (int i = t; i < n; i += 256) dist[i] = 1e10f;
  __syncthreads();
  int far = 0;
  for (int it = 0; it < S; ++it) {
    if (t == 0) out_idx[b * S + it] = far;
    float cx = px[far], cy = px[n + far], cz = px[2 * n + far];
    float best = -1.0f; int bestI = 0;
    for (int i = t; i < n; i += 256) {
      float dx = px[i] - cx, dy = px[n + i] - cy, dz = px[2 * n + i] - cz;
      float d = dx * dx + dy * dy + dz * dz;
      float dd = fminf(dist[i], d);
      dist[i] = dd;
      if (dd > best) { best = dd; bestI = i; }
    }
    rmax[t] = best; rarg[t] = bestI;
    __syncthreads();
    for (int s2 = 128; s2 > 0; s2 >>= 1) {
      if (t < s2 && rmax[t + s2] > rmax[t]) { rmax[t] = rmax[t + s2]; rarg[t] = rarg[t + s2]; }
      __syncthreads();
    }
    far = rarg[0];
    __syncthreads();
  }
}

// gather sampled coords: xyz [B,3,n], idx [B,S] -> nx [B,3,S]
__global__ void gather_xyz_kernel(const float* __restrict__ xyz, const int* __restrict__ idx,
                                  int B_, int n, int S, float* __restrict__ nx) {
  long long tot = (long long)B_ * 3 * S;
  GS(e, tot) {
    int s = (int)(e % S);
    long long t = e / S;
    int d = (int)(t % 3);
    int b = (int)(t / 3);
    nx[e] = xyz[((size_t)b * 3 + d) * n + idx[b * S + s]];
  }
}

// -------------------------------- ball query -------------------------------
__global__ void ball_query_kernel(const float* __restrict__ xyz, const float* __restrict__ nx,
                                  int B_, int n, int S, int K, float r2,
                                  int* __restrict__ gidx) {
  int g = blockIdx.x * blockDim.x + threadIdx.x;
  if (g >= B_ * S) return;
  int b = g / S, s = g % S;
  const float* px = xyz + (size_t)b * 3 * n;
  const float* pq = nx + (size_t)b * 3 * S;
  float qx = pq[s], qy = pq[S + s], qz = pq[2 * S + s];
  int* out = gidx + (size_t)g * K;
  int cnt = 0, first = 0; bool have = false;
  for (int j = 0; j < n && cnt < K; ++j) {
    float dx = px[j] - qx, dy = px[n + j] - qy, dz = px[2 * n + j] - qz;
    if (dx * dx + dy * dy + dz * dz <= r2) {
      if (!have) { first = j; have = true; }
      out[cnt++] = j;
    }
  }
  for (int j = cnt; j < K; ++j) out[j] = first;
}

// -------------------------------- grouping ---------------------------------
// build bf16 activation matrix X [C+3, B*S*K]; col = ((b*S+s)*K+k)
__global__ void group_kernel(const float* __restrict__ xyz, const float* __restrict__ pts,
                             const float* __restrict__ nx, const int* __restrict__ gidx,
                             int B_, int n, int S, int K, int C,
                             unsigned short* __restrict__ Xbf) {
  long long cols = (long long)B_ * S * K;
  long long tot = cols * (C + 3);
  GS(e, tot) {
    long long col = e % cols;
    int c = (int)(e / cols);
    long long bs = col / K;
    int s = (int)(bs % S);
    int b = (int)(bs / S);
    int j = gidx[col];
    float v;
    if (c < C) {
      v = pts[((size_t)b * C + c) * n + j];
    } else {
      int d = c - C;
      v = xyz[((size_t)b * 3 + d) * n + j] - nx[((size_t)b * 3 + d) * S + s];
    }
    Xbf[(size_t)c * cols + col] = f32_to_bf16_bits(v);
  }
}

// -------------------------------- WMMA GEMM --------------------------------
// Y[M, Ncols] = Wbf[Mpad,Kpad] * Xbf[Kpad, Ncols] + bias[M]
// Requirements (guaranteed by drivers): Kpad % 32 == 0, Ncols % 64 == 0,
// W zero-padded, X rows [K,Kpad) zeroed. 256 thr = 8 waves; tile 64x64;
// each wave: 1 M-subtile x 2 N-subtiles sharing one A fragment.
#define KT 32
#define LSTR 40   // LDS row stride in ushorts: 80B (16B multiple, conflict-free)

__global__ void __launch_bounds__(256)
gemm_wmma_kernel(const unsigned short* __restrict__ Wbf, const unsigned short* __restrict__ Xbf,
                 const float* __restrict__ bias, float* __restrict__ Y,
                 int M, int Kpad, long long Ncols) {
  __shared__ alignas(16) unsigned short Wt[64 * LSTR];   // [m][k]
  __shared__ alignas(16) unsigned short Xt[64 * LSTR];   // transposed: [n][k]
  int t = threadIdx.x;
  int wave = t >> 5, lane = t & 31;
  int mT = wave >> 1, nT = wave & 1;
  int half = lane >> 4, ml = lane & 15;
  long long n0 = (long long)blockIdx.x * 64;
  int m0 = blockIdx.y * 64;
  // staging tasks: one vector transfer per thread per K-step
  int wmm = t >> 2, wseg = t & 3;                        // W: 64 rows x 4 segs
  int xkk = t >> 3, xseg = t & 7;                        // X: 32 k x 8 n-segs
  const unsigned short* wsrc = Wbf + (size_t)(m0 + wmm) * Kpad + wseg * 8;
  const unsigned short* xsrc = Xbf + (size_t)xkk * Ncols + n0 + xseg * 8;
  v8f acc0 = {}, acc1 = {};
  for (int k0 = 0; k0 < Kpad; k0 += KT) {
    u16x8 wv = *(const u16x8*)(wsrc + k0);                      // aligned b128
    u16x8 xv = *(const u16x8*)(xsrc + (size_t)k0 * Ncols);      // aligned b128
    *(u16x8*)(Wt + wmm * LSTR + wseg * 8) = wv;                 // b128 LDS store
#pragma unroll
    for (int j = 0; j < 8; ++j)                                  // transpose scatter
      Xt[(xseg * 8 + j) * LSTR + xkk] = xv[j];
    if (k0 + KT < Kpad) {   // hint next K-slice into cache (global_prefetch_b8)
      __builtin_prefetch(xsrc + (size_t)(k0 + KT) * Ncols, 0, 1);
      __builtin_prefetch(wsrc + k0 + KT, 0, 1);
    }
    __syncthreads();
    // A fragment: lane half h holds K runs [8h..8h+7] and [16+8h..23+8h] of row M=ml
    const unsigned short* wrow = Wt + (mT * 16 + ml) * LSTR + half * 8;
    BF16Frag af;
    af.lo = *(const u16x8*)(wrow);
    af.hi = *(const u16x8*)(wrow + 16);
    v16bf a = __builtin_bit_cast(v16bf, af);
    // B fragments (X transposed => contiguous K runs per column)
    const unsigned short* xr0 = Xt + (nT * 32 + ml) * LSTR + half * 8;
    const unsigned short* xr1 = xr0 + 16 * LSTR;
    BF16Frag b0f, b1f;
    b0f.lo = *(const u16x8*)(xr0);
    b0f.hi = *(const u16x8*)(xr0 + 16);
    b1f.lo = *(const u16x8*)(xr1);
    b1f.hi = *(const u16x8*)(xr1 + 16);
    v16bf b0 = __builtin_bit_cast(v16bf, b0f);
    v16bf b1 = __builtin_bit_cast(v16bf, b1f);
    acc0 = __builtin_amdgcn_wmma_f32_16x16x32_bf16(false, a, false, b0, (short)0, acc0,
                                                   false, false);
    acc1 = __builtin_amdgcn_wmma_f32_16x16x32_bf16(false, a, false, b1, (short)0, acc1,
                                                   false, false);
    __syncthreads();
  }
  // C/D layout: VGPR v, lanes 0-15 -> M=v; lanes 16-31 -> M=v+8; N=lane&15
#pragma unroll
  for (int v = 0; v < 8; ++v) {
    int m = m0 + mT * 16 + v + 8 * half;
    if (m < M) {
      float bi = bias[m];
      long long nA = n0 + nT * 32 + ml;
      Y[(size_t)m * Ncols + nA] = acc0[v] + bi;
      Y[(size_t)m * Ncols + nA + 16] = acc1[v] + bi;
    }
  }
}

// -------------------------------- batch norm -------------------------------
__global__ void bn_stats_kernel(const float* __restrict__ Y, long long Ncols,
                                float* __restrict__ mean, float* __restrict__ var) {
  int c = blockIdx.x;
  const float* row = Y + (size_t)c * Ncols;
  __shared__ float s1[256], s2[256];
  float a = 0.f, b = 0.f;
  for (long long i = threadIdx.x; i < Ncols; i += 256) {
    float v = row[i]; a += v; b += v * v;
  }
  s1[threadIdx.x] = a; s2[threadIdx.x] = b;
  __syncthreads();
  for (int s = 128; s > 0; s >>= 1) {
    if (threadIdx.x < s) { s1[threadIdx.x] += s1[threadIdx.x + s]; s2[threadIdx.x] += s2[threadIdx.x + s]; }
    __syncthreads();
  }
  if (threadIdx.x == 0) {
    float m = s1[0] / (float)Ncols;
    mean[c] = m;
    var[c] = s2[0] / (float)Ncols - m * m;
  }
}

__global__ void bn_relu_kernel(const float* __restrict__ Y, const float* __restrict__ mean,
                               const float* __restrict__ var, const float* __restrict__ g,
                               const float* __restrict__ be, int M, long long Ncols,
                               float* __restrict__ Of32, unsigned short* __restrict__ Obf) {
  long long tot = (long long)M * Ncols;
  GS(e, tot) {
    int c = (int)(e / Ncols);
    float x = Y[e];
    float y = (x - mean[c]) * rsqrtf(var[c] + 1e-5f) * g[c] + be[c];
    y = fmaxf(y, 0.0f);
    Of32[e] = y;
    if (Obf) Obf[e] = f32_to_bf16_bits(y);
  }
}

// -------------------------------- max pool over K --------------------------
__global__ void maxpool_kernel(const float* __restrict__ Y, int cout, int B_, int S, int K,
                               float* __restrict__ pts_out, int Ctot, int coff) {
  long long cols = (long long)B_ * S * K;
  long long tot = (long long)B_ * cout * S;
  GS(e, tot) {
    int s = (int)(e % S);
    long long t = e / S;
    int o = (int)(t % cout);
    int b = (int)(t / cout);
    const float* row = Y + (size_t)o * cols + ((size_t)b * S + s) * K;
    float m = row[0];
    for (int k = 1; k < K; ++k) m = fmaxf(m, row[k]);
    pts_out[((size_t)b * Ctot + coff + o) * S + s] = m;
  }
}

// -------------------------------- FP: 3-NN + interp ------------------------
__global__ void fp_knn_kernel(const float* __restrict__ x1, const float* __restrict__ x2,
                              int B_, int n1, int n2, int* __restrict__ oi,
                              float* __restrict__ ow) {
  int g = blockIdx.x * blockDim.x + threadIdx.x;
  if (g >= B_ * n1) return;
  int b = g / n1, i = g % n1;
  const float* p1 = x1 + (size_t)b * 3 * n1;
  const float* p2 = x2 + (size_t)b * 3 * n2;
  float qx = p1[i], qy = p1[n1 + i], qz = p1[2 * n1 + i];
  float d0 = 3e38f, d1 = 3e38f, d2 = 3e38f;
  int i0 = 0, i1 = 0, i2 = 0;
  for (int j = 0; j < n2; ++j) {
    float dx = p2[j] - qx, dy = p2[n2 + j] - qy, dz = p2[2 * n2 + j] - qz;
    float d = dx * dx + dy * dy + dz * dz;
    if (d < d0)      { d2 = d1; i2 = i1; d1 = d0; i1 = i0; d0 = d; i0 = j; }
    else if (d < d1) { d2 = d1; i2 = i1; d1 = d;  i1 = j; }
    else if (d < d2) { d2 = d;  i2 = j; }
  }
  float w0 = 1.f / (d0 + 1e-8f), w1 = 1.f / (d1 + 1e-8f), w2 = 1.f / (d2 + 1e-8f);
  float s = w0 + w1 + w2;
  oi[g * 3] = i0; oi[g * 3 + 1] = i1; oi[g * 3 + 2] = i2;
  ow[g * 3] = w0 / s; ow[g * 3 + 1] = w1 / s; ow[g * 3 + 2] = w2 / s;
}

// build X [c1+c2, B*n1] bf16: rows 0..c1-1 = pts1, c1.. = interpolated pts2
__global__ void fp_build_kernel(const float* __restrict__ p1, const float* __restrict__ p2,
                                const int* __restrict__ oi, const float* __restrict__ ow,
                                int B_, int n1, int n2, int c1, int c2,
                                unsigned short* __restrict__ Xbf) {
  long long cols = (long long)B_ * n1;
  long long tot = cols * (c1 + c2);
  GS(e, tot) {
    long long col = e % cols;
    int c = (int)(e / cols);
    int i = (int)(col % n1);
    int b = (int)(col / n1);
    float v;
    if (c < c1) {
      v = p1[((size_t)b * c1 + c) * n1 + i];
    } else {
      int cc = c - c1;
      const int* id = oi + (size_t)col * 3;
      const float* w = ow + (size_t)col * 3;
      const float* row = p2 + ((size_t)b * c2 + cc) * n2;
      v = w[0] * row[id[0]] + w[1] * row[id[1]] + w[2] * row[id[2]];
    }
    Xbf[(size_t)c * cols + col] = f32_to_bf16_bits(v);
  }
}

// ---------------------------------------------------------------------------

struct LayerP {
  int cin, cout, kpad;
  const float *W, *b, *g, *be;
  unsigned short* Wbf;   // [pad64(cout) x kpad] zero-padded bf16
};

struct Bump {
  char* base; size_t off;
  void* take(size_t bytes) {
    size_t a = (off + 255) & ~(size_t)255;
    off = a + bytes;
    return base + a;
  }
};

extern "C" void kernel_launch(void* const* d_in, const int* in_sizes, int n_in,
                              void* d_out, int out_size, void* d_ws, size_t ws_size,
                              hipStream_t stream) {
  (void)in_sizes; (void)n_in; (void)out_size; (void)ws_size;
  const int B_ = 8, N0 = 4096;

  // ---------------- parameter table (traversal order of params dict) -------
  int pi = 1;
  auto nl = [&](int cin, int cout) -> LayerP {
    LayerP L;
    L.cin = cin; L.cout = cout; L.kpad = pad32(cin);
    L.W  = (const float*)d_in[pi + 0];
    L.b  = (const float*)d_in[pi + 1];
    L.g  = (const float*)d_in[pi + 2];
    L.be = (const float*)d_in[pi + 3];
    L.Wbf = nullptr;
    pi += 4;
    return L;
  };
  LayerP sa1s0[3] = { nl(9, 16),   nl(16, 16),   nl(16, 32) };
  LayerP sa1s1[3] = { nl(9, 32),   nl(32, 32),   nl(32, 64) };
  LayerP sa2s0[3] = { nl(99, 64),  nl(64, 64),   nl(64, 128) };
  LayerP sa2s1[3] = { nl(99, 64),  nl(64, 96),   nl(96, 128) };
  LayerP sa3s0[3] = { nl(259, 128), nl(128, 196), nl(196, 256) };
  LayerP sa3s1[3] = { nl(259, 128), nl(128, 196), nl(196, 256) };
  LayerP sa4s0[3] = { nl(515, 256), nl(256, 256), nl(256, 512) };
  LayerP sa4s1[3] = { nl(515, 256), nl(256, 384), nl(384, 512) };
  LayerP fp4L[2] = { nl(1536, 256), nl(256, 256) };
  LayerP fp3L[2] = { nl(512, 256),  nl(256, 256) };
  LayerP fp2L[2] = { nl(352, 256),  nl(256, 128) };
  LayerP conv1 = nl(128, 128);
  LayerP conv2 = nl(128, 128);
  const float* bng  = (const float*)d_in[pi++];
  const float* bnbe = (const float*)d_in[pi++];

  // ---------------- workspace carving --------------------------------------
  Bump al{ (char*)d_ws, 0 };
  float* xyz0 = (float*)al.take((size_t)B_ * 3 * N0 * 4);
  float* nx1  = (float*)al.take((size_t)B_ * 3 * 1024 * 4);
  float* pts1 = (float*)al.take((size_t)B_ * 96 * 1024 * 4);
  float* nx2  = (float*)al.take((size_t)B_ * 3 * 256 * 4);
  float* pts2 = (float*)al.take((size_t)B_ * 256 * 256 * 4);
  float* nx3  = (float*)al.take((size_t)B_ * 3 * 64 * 4);
  float* pts3 = (float*)al.take((size_t)B_ * 512 * 64 * 4);
  float* nx4  = (float*)al.take((size_t)B_ * 3 * 16 * 4);
  float* pts4 = (float*)al.take((size_t)B_ * 1024 * 16 * 4);
  float* l3p2 = (float*)al.take((size_t)B_ * 256 * 64 * 4);
  float* l2p2 = (float*)al.take((size_t)B_ * 256 * 256 * 4);
  float* l1p2 = (float*)al.take((size_t)B_ * 128 * 1024 * 4);
  int*   fps_idx = (int*)al.take((size_t)B_ * 1024 * 4);
  int*   gidx    = (int*)al.take((size_t)B_ * 1024 * 32 * 4);
  int*   knn_i   = (int*)al.take((size_t)B_ * 1024 * 3 * 4);
  float* knn_w   = (float*)al.take((size_t)B_ * 1024 * 3 * 4);
  float* meanb   = (float*)al.take(512 * 4);
  float* varb    = (float*)al.take(512 * 4);
  const long long ARENA = 16777216;  // 64 * 262144 (sa1 scale-2 worst case)
  float* Yar = (float*)al.take((size_t)ARENA * 4);
  unsigned short* Xar[2];
  Xar[0] = (unsigned short*)al.take((size_t)ARENA * 2);
  Xar[1] = (unsigned short*)al.take((size_t)ARENA * 2);

  auto nb = [](long long t) -> unsigned {
    long long b = (t + 255) / 256;
    if (b < 1) b = 1;
    if (b > 262144) b = 262144;
    return (unsigned)b;
  };
  auto zero_pad_rows = [&](unsigned short* X, int rows, int rpad, long long cols) {
    if (rpad > rows) {
      long long tot = (long long)(rpad - rows) * cols;
      zero_u16_kernel<<<nb(tot), 256, 0, stream>>>(X + (size_t)rows * cols, tot);
    }
  };

  // ---------------- weight prep: bf16, [pad64(M) x pad32(K)] ---------------
  LayerP* groups[] = { sa1s0, sa1s1, sa2s0, sa2s1, sa3s0, sa3s1, sa4s0, sa4s1,
                       fp4L, fp3L, fp2L, &conv1, &conv2 };
  int counts[] = { 3, 3, 3, 3, 3, 3, 3, 3, 2, 2, 2, 1, 1 };
  for (int gI = 0; gI < 13; ++gI) {
    for (int li = 0; li < counts[gI]; ++li) {
      LayerP& L = groups[gI][li];
      long long tot = (long long)pad64(L.cout) * L.kpad;
      L.Wbf = (unsigned short*)al.take((size_t)tot * 2);
      w_prep_kernel<<<nb(tot), 256, 0, stream>>>(L.W, L.cout, L.cin, L.kpad, L.Wbf, tot);
    }
  }

  // ---------------- MLP stack driver ---------------------------------------
  auto mlp = [&](LayerP* Ls, int nlr, long long cols, int cur) -> float* {
    for (int l = 0; l < nlr; ++l) {
      LayerP& L = Ls[l];
      dim3 g((unsigned)((cols + 63) / 64), (unsigned)((L.cout + 63) / 64));
      gemm_wmma_kernel<<<g, 256, 0, stream>>>(L.Wbf, Xar[cur], L.b, Yar, L.cout, L.kpad, cols);
      bn_stats_kernel<<<L.cout, 256, 0, stream>>>(Yar, cols, meanb, varb);
      unsigned short* obf = (l + 1 < nlr) ? Xar[1 - cur] : nullptr;
      long long tot = (long long)L.cout * cols;
      bn_relu_kernel<<<nb(tot), 256, 0, stream>>>(Yar, meanb, varb, L.g, L.be, L.cout, cols,
                                                  Yar, obf);
      if (obf) zero_pad_rows(obf, L.cout, Ls[l + 1].kpad, cols);
      cur = 1 - cur;
    }
    return Yar;
  };

  // ---------------- SA (set abstraction, MSG) ------------------------------
  auto run_sa = [&](const float* xyzL, const float* ptsL, int n, int C, int S,
                    float r0, int K0, LayerP* s0, float r1, int K1, LayerP* s1,
                    float* nxo, float* ptso, int Ctot) {
    fps_kernel<<<B_, 256, (size_t)n * 4 + 256 * 8, stream>>>(xyzL, n, S, fps_idx);
    gather_xyz_kernel<<<nb((long long)B_ * 3 * S), 256, 0, stream>>>(xyzL, fps_idx, B_, n, S, nxo);
    auto scale = [&](float r, int K, LayerP* Ls, int coff) {
      ball_query_kernel<<<(B_ * S + 127) / 128, 128, 0, stream>>>(xyzL, nxo, B_, n, S, K,
                                                                  r * r, gidx);
      long long cols = (long long)B_ * S * K;
      group_kernel<<<nb(cols * (C + 3)), 256, 0, stream>>>(xyzL, ptsL, nxo, gidx, B_, n, S, K,
                                                           C, Xar[0]);
      zero_pad_rows(Xar[0], C + 3, Ls[0].kpad, cols);
      float* Y = mlp(Ls, 3, cols, 0);
      maxpool_kernel<<<nb((long long)B_ * Ls[2].cout * S), 256, 0, stream>>>(
          Y, Ls[2].cout, B_, S, K, ptso, Ctot, coff);
    };
    scale(r0, K0, s0, 0);
    scale(r1, K1, s1, s0[2].cout);
  };

  // ---------------- FP (feature propagation) -------------------------------
  auto run_fp = [&](const float* x1, const float* x2, const float* p1, const float* p2,
                    int n1, int n2, int c1, int c2, LayerP* Ls, float* ptso) {
    fp_knn_kernel<<<(B_ * n1 + 127) / 128, 128, 0, stream>>>(x1, x2, B_, n1, n2, knn_i, knn_w);
    long long cols = (long long)B_ * n1;
    fp_build_kernel<<<nb(cols * (c1 + c2)), 256, 0, stream>>>(p1, p2, knn_i, knn_w, B_, n1, n2,
                                                              c1, c2, Xar[0]);
    zero_pad_rows(Xar[0], c1 + c2, Ls[0].kpad, cols);
    float* Y = mlp(Ls, 2, cols, 0);
    rows_to_bcs_kernel<<<nb((long long)B_ * Ls[1].cout * n1), 256, 0, stream>>>(
        Y, Ls[1].cout, B_, n1, ptso);
  };

  // ---------------- network ------------------------------------------------
  const float* xyz_in = (const float*)d_in[0];   // [B, 6, 4096]
  slice_channels_kernel<<<nb((long long)B_ * 3 * N0), 256, 0, stream>>>(xyz_in, B_, 6, N0, 0, 3,
                                                                        xyz0);

  run_sa(xyz0, xyz_in, N0, 6, 1024, 0.05f, 16, sa1s0, 0.10f, 32, sa1s1, nx1, pts1, 96);
  run_sa(nx1, pts1, 1024, 96, 256, 0.10f, 16, sa2s0, 0.20f, 32, sa2s1, nx2, pts2, 256);
  run_sa(nx2, pts2, 256, 256, 64, 0.20f, 16, sa3s0, 0.40f, 32, sa3s1, nx3, pts3, 512);
  run_sa(nx3, pts3, 64, 512, 16, 0.40f, 16, sa4s0, 0.80f, 32, sa4s1, nx4, pts4, 1024);

  run_fp(nx3, nx4, pts3, pts4, 64, 16, 512, 1024, fp4L, l3p2);
  run_fp(nx2, nx3, pts2, l3p2, 256, 64, 256, 256, fp3L, l2p2);
  run_fp(nx1, nx2, pts1, l2p2, 1024, 256, 96, 256, fp2L, l1p2);

  // head: conv1 -> bn(params['bn']) -> relu -> conv2
  {
    long long cols = (long long)B_ * 1024;
    bcs_to_rows_bf16_kernel<<<nb(128 * cols), 256, 0, stream>>>(l1p2, 128, B_, 1024, Xar[0]);
    dim3 g((unsigned)((cols + 63) / 64), 2);
    gemm_wmma_kernel<<<g, 256, 0, stream>>>(conv1.Wbf, Xar[0], conv1.b, Yar, 128, conv1.kpad,
                                            cols);
    bn_stats_kernel<<<128, 256, 0, stream>>>(Yar, cols, meanb, varb);
    bn_relu_kernel<<<nb(128 * cols), 256, 0, stream>>>(Yar, meanb, varb, bng, bnbe, 128, cols,
                                                       Yar, Xar[1]);
    gemm_wmma_kernel<<<g, 256, 0, stream>>>(conv2.Wbf, Xar[1], conv2.b, Yar, 128, conv2.kpad,
                                            cols);
    rows_to_bcs_kernel<<<nb(128 * cols), 256, 0, stream>>>(
        Yar, 128, B_, 1024, (float*)d_out + (size_t)B_ * 3 * 1024);
    copy_f32_kernel<<<nb((long long)B_ * 3 * 1024), 256, 0, stream>>>(
        nx1, (float*)d_out, (long long)B_ * 3 * 1024);
  }
}
